// TtLlamaAttention_58085137711159
// MI455X (gfx1250) — matrix operand order repaired and checked by reading, not compiled, not run on hardware
//
#include <hip/hip_runtime.h>
#include <hip/hip_bf16.h>
#include <stdint.h>

// ---------------------------------------------------------------------------
// Types / helpers
// ---------------------------------------------------------------------------
typedef __bf16 bf16;
typedef __attribute__((ext_vector_type(16))) __bf16 v16bf;
typedef __attribute__((ext_vector_type(8)))  float  v8f;

union Frag16 { v16bf v; uint4 q[2]; };

static __device__ __forceinline__ bf16 f2bf(float f) {
    unsigned u = __builtin_bit_cast(unsigned, f);
    u += 0x7FFFu + ((u >> 16) & 1u);            // round-to-nearest-even
    unsigned short h = (unsigned short)(u >> 16);
    return __builtin_bit_cast(bf16, h);
}
static __device__ __forceinline__ float bf2f(bf16 b) {
    unsigned u = ((unsigned)__builtin_bit_cast(unsigned short, b)) << 16;
    return __builtin_bit_cast(float, u);
}

static __device__ __forceinline__ float rmax16(float x) {
    // butterfly over the 16-lane half (xor masks 8,4,2,1 stay inside each half)
    #pragma unroll
    for (int m = 8; m >= 1; m >>= 1) x = fmaxf(x, __shfl_xor(x, m, 32));
    return x;
}
static __device__ __forceinline__ float rsum16(float x) {
    #pragma unroll
    for (int m = 8; m >= 1; m >>= 1) x += __shfl_xor(x, m, 32);
    return x;
}

static __device__ __forceinline__ v8f wmma_bf16(const v16bf& a, const v16bf& b, const v8f& c) {
    return __builtin_amdgcn_wmma_f32_16x16x32_bf16(
        /*neg_a=*/false, a, /*neg_b=*/false, b,
        /*c_mod=*/(short)0, c, /*reuse_a=*/false, /*reuse_b=*/false);
}

// Async memory->LDS copy of 16B per lane (CDNA5 ASYNCcnt path).
// ldsoff = low 32 bits of the generic shared-memory pointer (== LDS byte
// address per the flat->LDS aperture truncation rule, ISA 10.2).
static __device__ __forceinline__ void async_g2l_b128(unsigned ldsoff, const void* gptr) {
    asm volatile("global_load_async_to_lds_b128 %0, %1, off"
                 :: "v"(ldsoff), "v"(gptr) : "memory");
}
static __device__ __forceinline__ void wait_async0() {
    asm volatile("s_wait_asynccnt 0" ::: "memory");
}

// Problem constants
#define S_LEN   2048
#define HID     4096
#define NHEAD   32
#define HDIM    128
#define SCALE   0.08838834764831845f   // 1/sqrt(128)

// ---------------------------------------------------------------------------
// Kernel 1: fp32 -> bf16 convert (vectorized x4)
// ---------------------------------------------------------------------------
__global__ __launch_bounds__(256) void k_cvt_bf16(const float* __restrict__ src,
                                                  bf16* __restrict__ dst, int n4) {
    int i = blockIdx.x * blockDim.x + threadIdx.x;
    if (i >= n4) return;
    float4 f = ((const float4*)src)[i];
    unsigned lo = (unsigned)__builtin_bit_cast(unsigned short, f2bf(f.x)) |
                  ((unsigned)__builtin_bit_cast(unsigned short, f2bf(f.y)) << 16);
    unsigned hi = (unsigned)__builtin_bit_cast(unsigned short, f2bf(f.z)) |
                  ((unsigned)__builtin_bit_cast(unsigned short, f2bf(f.w)) << 16);
    ((uint2*)dst)[i] = make_uint2(lo, hi);
}

// ---------------------------------------------------------------------------
// Kernel 2: GEMM  C[M,N] = A[M,K](bf16) * B[N,K](bf16)^T, fp32 accum.
// Block = 256 threads (8 waves), block tile 128x128, wave tile 16x128.
// B staged in LDS with DOUBLE-BUFFERED GLOBAL_LOAD_ASYNC_TO_LDS_B128:
// while tile i feeds 16 WMMAs, the async unit streams tile i+1 into the
// other LDS buffer (ASYNCcnt overlap, no VGPR involvement).
// ---------------------------------------------------------------------------
#define KC       64          // k-extent staged per pipeline stage
#define BSTRIDE2 72          // padded LDS row (elements): conflict-free b128 reads
template <bool F32OUT>
__global__ __launch_bounds__(256) void k_gemm_nt(const bf16* __restrict__ A,
                                                 const bf16* __restrict__ B,
                                                 void* __restrict__ Cout,
                                                 int M, int N, int K) {
    __shared__ bf16 Blds[2][128][BSTRIDE2];
    const int lane = threadIdx.x & 31;
    const int wave = threadIdx.x >> 5;
    const int n0 = blockIdx.x * 128;
    const int m0 = blockIdx.y * 128 + wave * 16;

    v8f acc[8];
    const v8f zv = {0.f, 0.f, 0.f, 0.f, 0.f, 0.f, 0.f, 0.f};
    #pragma unroll
    for (int nf = 0; nf < 8; ++nf) acc[nf] = zv;

    const int aRow   = m0 + (lane & 15);
    const int aChunk = (lane >> 4) << 3;     // 0 or 8 (ISA 16-bit A layout)
    const int bCol   = lane & 15;
    const int bK     = (lane >> 4) << 3;     // 0 or 8 (ISA 16-bit B layout)

    // Stage one 128 x KC tile: 1024 x 16B transfers, 4 per thread.
    auto stage = [&](int buf, int kbase) {
        #pragma unroll
        for (int i = 0; i < 4; ++i) {
            int idx = threadIdx.x + i * 256;       // 0..1023
            int n   = idx >> 3;                    // 0..127
            int kq  = (idx & 7) << 3;              // 0,8,...,56
            async_g2l_b128((unsigned)(uintptr_t)&Blds[buf][n][kq],
                           B + (size_t)(n0 + n) * K + kbase + kq);
        }
    };

    stage(0, 0);                                   // prologue
    int buf = 0;
    for (int k0 = 0; k0 < K; k0 += KC, buf ^= 1) {
        wait_async0();                             // tile `buf` landed in LDS
        __syncthreads();
        if (k0 + KC < K) stage(buf ^ 1, k0 + KC);  // overlap next tile copy

        #pragma unroll
        for (int h = 0; h < 2; ++h) {              // two k=32 half-steps
            Frag16 a;
            const bf16* ap = A + (size_t)aRow * K + k0 + h * 32 + aChunk;
            a.q[0] = *(const uint4*)ap;            // K {c..c+7}
            a.q[1] = *(const uint4*)(ap + 16);     // K {c+16..c+23}

            #pragma unroll
            for (int nf = 0; nf < 8; ++nf) {
                Frag16 b;
                const bf16* bp = &Blds[buf][nf * 16 + bCol][h * 32 + bK];
                b.q[0] = *(const uint4*)bp;
                b.q[1] = *(const uint4*)(bp + 16);
                acc[nf] = wmma_bf16(a.v, b.v, acc[nf]);
            }
        }
        __syncthreads();                           // all reads of `buf` done
    }

    // C layout: N = n0+nf*16+(lane&15), M = m0 + r + 8*(lane>>4)
    const int cc    = lane & 15;
    const int rbase = (lane >> 4) << 3;
    #pragma unroll
    for (int nf = 0; nf < 8; ++nf) {
        #pragma unroll
        for (int r = 0; r < 8; ++r) {
            size_t off = (size_t)(m0 + rbase + r) * N + n0 + nf * 16 + cc;
            if (F32OUT) ((float*)Cout)[off] = acc[nf][r];
            else        ((bf16*)Cout)[off]  = f2bf(acc[nf][r]);
        }
    }
}

// ---------------------------------------------------------------------------
// Kernel 3: RoPE on Q and K (in place, bf16 storage, fp32 math).
// One thread owns pair (d, d+64) of one (s, head).
// ---------------------------------------------------------------------------
__global__ __launch_bounds__(256) void k_rope(bf16* __restrict__ Q, bf16* __restrict__ K,
                                              const long long* __restrict__ pos_ids) {
    int idx = blockIdx.x * blockDim.x + threadIdx.x;   // 2048*32*64 threads
    int s    = idx >> 11;
    int rem  = idx & 2047;
    int head = rem >> 6;
    int d    = rem & 63;
    float pos  = (float)pos_ids[s];
    float invf = __expf(-(2.0f * d / 128.0f) * 9.2103403719761836f); // ln(1e4)
    float ang  = pos * invf;
    float c = __cosf(ang), sn = __sinf(ang);
    size_t base = (size_t)s * HID + head * HDIM + d;
    float q1 = bf2f(Q[base]), q2 = bf2f(Q[base + 64]);
    float k1 = bf2f(K[base]), k2 = bf2f(K[base + 64]);
    Q[base]      = f2bf(q1 * c - q2 * sn);
    Q[base + 64] = f2bf(q2 * c + q1 * sn);
    K[base]      = f2bf(k1 * c - k2 * sn);
    K[base + 64] = f2bf(k2 * c + k1 * sn);
}

// ---------------------------------------------------------------------------
// Kernel 4: V [s][head*128+d] -> Vt [head][d][s]  (bf16)
// ---------------------------------------------------------------------------
__global__ __launch_bounds__(256) void k_vtrans(const bf16* __restrict__ V,
                                                bf16* __restrict__ Vt) {
    int idx = blockIdx.x * blockDim.x + threadIdx.x;   // 32*128*2048 threads
    int s    = idx & (S_LEN - 1);
    int d    = (idx >> 11) & (HDIM - 1);
    int head = idx >> 18;
    Vt[idx] = V[(size_t)s * HID + head * HDIM + d];
}

// ---------------------------------------------------------------------------
// Kernel 5: causal flash attention, one wave = 16-query strip, head per block.x
// ---------------------------------------------------------------------------
#define PSTRIDE 40
__global__ __launch_bounds__(256) void k_attn(const bf16* __restrict__ Q,
                                              const bf16* __restrict__ Kr,
                                              const bf16* __restrict__ Vt,
                                              bf16* __restrict__ Ctx) {
    __shared__ bf16 Pl[8][16][PSTRIDE];     // per-wave 16x32 P scratch (padded)
    const int lane = threadIdx.x & 31;
    const int wave = threadIdx.x >> 5;
    const int head = blockIdx.x;
    const int q0   = blockIdx.y * 128 + wave * 16;

    const int cc    = lane & 15;
    const int rbase = (lane >> 4) << 3;

    // Q A-fragments: 4 k-chunks of 32 covering d=0..127
    Frag16 qf[4];
    {
        const int aChunk = (lane >> 4) << 3;
        #pragma unroll
        for (int kc = 0; kc < 4; ++kc) {
            const bf16* qp = Q + (size_t)(q0 + cc) * HID + head * HDIM + kc * 32 + aChunk;
            qf[kc].q[0] = *(const uint4*)qp;
            qf[kc].q[1] = *(const uint4*)(qp + 16);
        }
    }

    v8f o[8];
    const v8f zv = {0.f, 0.f, 0.f, 0.f, 0.f, 0.f, 0.f, 0.f};
    #pragma unroll
    for (int nf = 0; nf < 8; ++nf) o[nf] = zv;
    float mi[8], li[8];
    #pragma unroll
    for (int r = 0; r < 8; ++r) { mi[r] = -1e30f; li[r] = 0.f; }

    const int kmax = q0 + 15;
    for (int k0 = 0; k0 <= kmax; k0 += 32) {
        float sv[2][8];
        #pragma unroll
        for (int j = 0; j < 2; ++j) {
            v8f s = zv;
            const int key = k0 + j * 16 + cc;
            #pragma unroll
            for (int kc = 0; kc < 4; ++kc) {
                Frag16 kf;
                const bf16* kp = Kr + (size_t)key * HID + head * HDIM + kc * 32 + ((lane >> 4) << 3);
                kf.q[0] = *(const uint4*)kp;
                kf.q[1] = *(const uint4*)(kp + 16);
                s = wmma_bf16(qf[kc].v, kf.v, s);
            }
            #pragma unroll
            for (int r = 0; r < 8; ++r) {
                int qrow = q0 + rbase + r;
                sv[j][r] = (key <= qrow) ? s[r] * SCALE : -1e30f;
            }
        }
        // online softmax over the 32-key chunk
        float alpha[8];
        #pragma unroll
        for (int r = 0; r < 8; ++r) {
            float mx = rmax16(fmaxf(sv[0][r], sv[1][r]));
            float nm = fmaxf(mi[r], mx);
            alpha[r] = __expf(mi[r] - nm);
            mi[r] = nm;
        }
        #pragma unroll
        for (int r = 0; r < 8; ++r) {
            sv[0][r] = __expf(sv[0][r] - mi[r]);
            sv[1][r] = __expf(sv[1][r] - mi[r]);
            li[r] = li[r] * alpha[r] + rsum16(sv[0][r] + sv[1][r]);
        }
        #pragma unroll
        for (int nf = 0; nf < 8; ++nf)
            #pragma unroll
            for (int r = 0; r < 8; ++r) o[nf][r] *= alpha[r];

        // P (C-layout) -> LDS -> A-layout fragment
        #pragma unroll
        for (int j = 0; j < 2; ++j)
            #pragma unroll
            for (int r = 0; r < 8; ++r)
                Pl[wave][rbase + r][j * 16 + cc] = f2bf(sv[j][r]);
        asm volatile("s_wait_dscnt 0" ::: "memory");
        Frag16 pf;
        {
            const bf16* pp = &Pl[wave][cc][(lane >> 4) << 3];
            pf.q[0] = *(const uint4*)pp;
            pf.q[1] = *(const uint4*)(pp + 16);
        }
        // P @ V  (Vt: [head][d][s], contiguous along keys)
        #pragma unroll
        for (int nf = 0; nf < 8; ++nf) {
            Frag16 vf;
            const bf16* vp = Vt + ((size_t)head * HDIM + nf * 16 + cc) * S_LEN + k0 + ((lane >> 4) << 3);
            vf.q[0] = *(const uint4*)vp;
            vf.q[1] = *(const uint4*)(vp + 16);
            o[nf] = wmma_bf16(pf.v, vf.v, o[nf]);
        }
    }

    // normalize and write context [s][head*128+d] as bf16
    #pragma unroll
    for (int r = 0; r < 8; ++r) {
        float inv = 1.0f / li[r];
        #pragma unroll
        for (int nf = 0; nf < 8; ++nf)
            Ctx[(size_t)(q0 + rbase + r) * HID + head * HDIM + nf * 16 + cc] = f2bf(o[nf][r] * inv);
    }
}

// ---------------------------------------------------------------------------
// Launcher
// ---------------------------------------------------------------------------
extern "C" void kernel_launch(void* const* d_in, const int* in_sizes, int n_in,
                              void* d_out, int out_size, void* d_ws, size_t ws_size,
                              hipStream_t stream) {
    (void)in_sizes; (void)n_in; (void)out_size; (void)ws_size;
    const float*      x    = (const float*)d_in[0];
    // d_in[1] = attention_mask (causal; applied analytically)
    const float*      Wq   = (const float*)d_in[2];
    const float*      Wk   = (const float*)d_in[3];
    const float*      Wv   = (const float*)d_in[4];
    const float*      Wo   = (const float*)d_in[5];
    const long long*  pos  = (const long long*)d_in[6];
    float*            out  = (float*)d_out;

    char* ws = (char*)d_ws;
    const size_t XBYTES = (size_t)S_LEN * HID * 2;       // 16 MiB
    const size_t WBYTES = (size_t)HID * HID * 2;         // 32 MiB
    bf16* xb   = (bf16*)(ws);
    bf16* wqb  = (bf16*)(ws + XBYTES);
    bf16* wkb  = (bf16*)(ws + XBYTES + WBYTES);
    bf16* wvb  = (bf16*)(ws + XBYTES + 2 * WBYTES);
    bf16* wob  = (bf16*)(ws + XBYTES + 3 * WBYTES);
    bf16* qraw = (bf16*)(ws + XBYTES + 4 * WBYTES);
    bf16* kraw = (bf16*)(ws + 2 * XBYTES + 4 * WBYTES);
    bf16* vraw = (bf16*)(ws + 3 * XBYTES + 4 * WBYTES);
    bf16* vt   = (bf16*)(ws + 4 * XBYTES + 4 * WBYTES);
    bf16* ctx  = (bf16*)(ws + 5 * XBYTES + 4 * WBYTES);

    const int NX4 = S_LEN * HID / 4;       // 2,097,152
    const int NW4 = HID * HID / 4;         // 4,194,304
    k_cvt_bf16<<<(NX4 + 255) / 256, 256, 0, stream>>>(x,  xb,  NX4);
    k_cvt_bf16<<<(NW4 + 255) / 256, 256, 0, stream>>>(Wq, wqb, NW4);
    k_cvt_bf16<<<(NW4 + 255) / 256, 256, 0, stream>>>(Wk, wkb, NW4);
    k_cvt_bf16<<<(NW4 + 255) / 256, 256, 0, stream>>>(Wv, wvb, NW4);
    k_cvt_bf16<<<(NW4 + 255) / 256, 256, 0, stream>>>(Wo, wob, NW4);

    dim3 ggrid(HID / 128, S_LEN / 128);    // (32, 16)
    k_gemm_nt<false><<<ggrid, 256, 0, stream>>>(xb, wqb, qraw, S_LEN, HID, HID);
    k_gemm_nt<false><<<ggrid, 256, 0, stream>>>(xb, wkb, kraw, S_LEN, HID, HID);
    k_gemm_nt<false><<<ggrid, 256, 0, stream>>>(xb, wvb, vraw, S_LEN, HID, HID);

    k_rope<<<(S_LEN * NHEAD * 64) / 256, 256, 0, stream>>>(qraw, kraw, pos);
    k_vtrans<<<(S_LEN * HID) / 256, 256, 0, stream>>>(vraw, vt);

    k_attn<<<dim3(NHEAD, S_LEN / 128), 256, 0, stream>>>(qraw, kraw, vt, ctx);

    k_gemm_nt<true><<<ggrid, 256, 0, stream>>>(ctx, wob, out, S_LEN, HID, HID);
}
